// GCN_success_51694226375265
// MI455X (gfx1250) — matrix operand order repaired and checked by reading, not compiled, not run on hardware
//
#include <hip/hip_runtime.h>
#include <hip/hip_bf16.h>

// ---- problem constants (from reference) ----
#define Nn   20000
#define Ee   320000
#define EP   340000      // E + N self loops
#define Hh   6
#define Cc   64
#define HC   384
#define Gg   64
#define NF   6
#define NT   4           // 16-wide N-tiles per wave in the WMMA GEMM

typedef __attribute__((ext_vector_type(8)))  __bf16 v8bf;
typedef __attribute__((ext_vector_type(16))) __bf16 v16bf;
typedef __attribute__((ext_vector_type(8)))  float  v8f;

// ---------- helpers ----------
__device__ __forceinline__ unsigned enc_f(float f) {
    unsigned u = __float_as_uint(f);
    return (u & 0x80000000u) ? ~u : (u | 0x80000000u);
}
__device__ __forceinline__ float dec_f(unsigned u) {
    return __uint_as_float((u & 0x80000000u) ? (u & 0x7FFFFFFFu) : ~u);
}
__device__ __forceinline__ unsigned short f2bf(float f) {
    unsigned u = __float_as_uint(f);
    u += 0x7FFFu + ((u >> 16) & 1u);     // round-to-nearest-even
    return (unsigned short)(u >> 16);
}

// ---------- tiny utility kernels ----------
__global__ void k_fill_u32(unsigned* p, unsigned v, int n) {
    int i = blockIdx.x * blockDim.x + threadIdx.x;
    if (i < n) p[i] = v;
}

__global__ void k_mean_sum(const float* __restrict__ ea, float* sum) {
    __shared__ float sh[256];
    float acc = 0.f;
    for (int i = blockIdx.x * blockDim.x + threadIdx.x; i < Ee; i += gridDim.x * blockDim.x)
        acc += ea[i];
    sh[threadIdx.x] = acc;
    __syncthreads();
    for (int s = 128; s > 0; s >>= 1) {
        if (threadIdx.x < s) sh[threadIdx.x] += sh[threadIdx.x + s];
        __syncthreads();
    }
    if (threadIdx.x == 0) atomicAdd(sum, sh[0]);
}

// coef[h] = sum_c We[h*64+c] * AttE[h*64+c]   (EDGE_DIM==1)
__global__ void k_coef(const float* __restrict__ we1, const float* __restrict__ ae1,
                       const float* __restrict__ we2, const float* __restrict__ ae2,
                       float* __restrict__ coef) {
    int t = threadIdx.x;
    if (t >= 2 * Hh) return;
    const float* w = (t < Hh) ? we1 : we2;
    const float* a = (t < Hh) ? ae1 : ae2;
    int h = (t < Hh) ? t : (t - Hh);
    float s = 0.f;
    for (int c = 0; c < Cc; ++c) s += w[h * Cc + c] * a[h * Cc + c];
    coef[t] = s;
}

// ---------- conv1 input projection: h = x @ W1^T  (K = 6, scalar) ----------
__global__ void k_proj1(const float* __restrict__ x, const float* __restrict__ w1,
                        float* __restrict__ h) {
    long idx = (long)blockIdx.x * blockDim.x + threadIdx.x;
    if (idx >= (long)Nn * HC) return;
    int n = (int)(idx / HC), j = (int)(idx % HC);
    float s = 0.f;
    #pragma unroll
    for (int k = 0; k < NF; ++k) s += x[n * NF + k] * w1[j * NF + k];
    h[idx] = s;
}

// ---------- per-node attention scores: a_s[n,h], a_d[n,h] ----------
__global__ void k_attn_scores(const float* __restrict__ h,
                              const float* __restrict__ att_src,
                              const float* __restrict__ att_dst,
                              float* __restrict__ as_, float* __restrict__ ad_) {
    int t = blockIdx.x * blockDim.x + threadIdx.x;
    if (t >= Nn * Hh) return;
    int n = t / Hh, hh = t % Hh;
    const float4* hv = (const float4*)(h + (size_t)n * HC + hh * Cc);
    const float4* sv = (const float4*)(att_src + hh * Cc);
    const float4* dv = (const float4*)(att_dst + hh * Cc);
    float ss = 0.f, sd = 0.f;
    #pragma unroll
    for (int c = 0; c < Cc / 4; ++c) {
        float4 a = hv[c], b = sv[c], d = dv[c];
        ss += a.x * b.x + a.y * b.y + a.z * b.z + a.w * b.w;
        sd += a.x * d.x + a.y * d.y + a.z * d.z + a.w * d.w;
    }
    as_[t] = ss;
    ad_[t] = sd;
}

// ---------- edge pass A: alpha = leaky_relu(...), atomicMax per (dst,h) ----------
__global__ void k_edge_alpha(const int* __restrict__ srcI, const int* __restrict__ dstI,
                             const float* __restrict__ ea, const float* __restrict__ easum,
                             const float* __restrict__ coef,
                             const float* __restrict__ as_, const float* __restrict__ ad_,
                             float* __restrict__ alpha, unsigned* __restrict__ nmax) {
    long t = (long)blockIdx.x * blockDim.x + threadIdx.x;
    if (t >= (long)EP * Hh) return;
    int e = (int)(t / Hh), hh = (int)(t % Hh);
    int s, d; float eav;
    if (e < Ee) { s = srcI[e]; d = dstI[e]; eav = ea[e]; }
    else        { s = e - Ee;  d = s;       eav = easum[0] * (1.0f / (float)Ee); }
    float a = as_[s * Hh + hh] + ad_[d * Hh + hh] + coef[hh] * eav;
    a = (a > 0.f) ? a : 0.2f * a;                 // leaky_relu(0.2)
    alpha[t] = a;
    atomicMax(&nmax[d * Hh + hh], enc_f(a));
}

// ---------- edge pass B: ex = exp(alpha - max[dst]); atomicAdd denom ----------
__global__ void k_edge_exp(const int* __restrict__ srcI, const int* __restrict__ dstI,
                           float* __restrict__ alpha, const unsigned* __restrict__ nmax,
                           float* __restrict__ nden) {
    long t = (long)blockIdx.x * blockDim.x + threadIdx.x;
    if (t >= (long)EP * Hh) return;
    int e = (int)(t / Hh), hh = (int)(t % Hh);
    int d = (e < Ee) ? dstI[e] : (e - Ee);
    float ex = __expf(alpha[t] - dec_f(nmax[d * Hh + hh]));
    alpha[t] = ex;
    atomicAdd(&nden[d * Hh + hh], ex);
}

// ---------- edge pass C: out[dst] += h[src] * att ; 4 channels per thread ----------
__global__ void k_edge_scatter(const int* __restrict__ srcI, const int* __restrict__ dstI,
                               const float* __restrict__ ex, const float* __restrict__ nden,
                               const float* __restrict__ h, float* __restrict__ out) {
    long t = (long)blockIdx.x * blockDim.x + threadIdx.x;
    if (t >= (long)EP * Hh * (Cc / 4)) return;
    int c4 = (int)(t & 15);
    long r = t >> 4;
    int hh = (int)(r % Hh);
    int e  = (int)(r / Hh);
    int s, d;
    if (e < Ee) { s = srcI[e]; d = dstI[e]; }
    else        { s = e - Ee;  d = s; }
    float att = ex[(long)e * Hh + hh] / nden[d * Hh + hh];
    const float4 hv = *(const float4*)(h + (size_t)s * HC + hh * Cc + c4 * 4);
    float* o = out + (size_t)d * HC + hh * Cc + c4 * 4;
    atomicAdd(o + 0, hv.x * att);
    atomicAdd(o + 1, hv.y * att);
    atomicAdd(o + 2, hv.z * att);
    atomicAdd(o + 3, hv.w * att);
}

// ---------- bias + ELU (in place) ----------
__global__ void k_bias_elu(float* __restrict__ x, const float* __restrict__ b) {
    long idx = (long)blockIdx.x * blockDim.x + threadIdx.x;
    if (idx >= (long)Nn * HC) return;
    float v = x[idx] + b[(int)(idx % HC)];
    x[idx] = (v > 0.f) ? v : (__expf(v) - 1.0f);
}

// ---------- f32 -> bf16 elementwise ----------
__global__ void k_f2bf(const float* __restrict__ s, unsigned short* __restrict__ d, long n) {
    long i = (long)blockIdx.x * blockDim.x + threadIdx.x;
    if (i < n) d[i] = f2bf(s[i]);
}

// ---------- bf16 WMMA GEMM: D[M,384] = A[M,384] * B[384,384]^T ----------
// one wave per 16 x (16*NT) strip; A-fragment reused across the NT n-tiles.
__global__ __launch_bounds__(32)
void k_wmma_gemm(const unsigned short* __restrict__ A, const unsigned short* __restrict__ B,
                 float* __restrict__ D) {
    const int m0 = blockIdx.x * 16;
    const int n0 = blockIdx.y * (16 * NT);
    const int lane = threadIdx.x;
    const int half = lane >> 4;      // 0: lanes 0-15, 1: lanes 16-31
    const int l    = lane & 15;

    v8f acc[NT];
    #pragma unroll
    for (int j = 0; j < NT; ++j) acc[j] = (v8f){};

    const unsigned short* arow = A + (size_t)(m0 + l) * HC;          // A row m0+l
    const unsigned short* brow[NT];
    #pragma unroll
    for (int j = 0; j < NT; ++j)
        brow[j] = B + (size_t)(n0 + j * 16 + l) * HC;                // B col == W2 row

    for (int k0 = 0; k0 < HC; k0 += 32) {
        // A-fragment, 16-bit 16x32 layout: lane-half selects K-offset 0/8;
        // elems 0..7 = K k0+half*8+0..7, elems 8..15 = K k0+16+half*8+0..7
        union { v16bf v; v8bf p[2]; } af;
        af.p[0] = *(const v8bf*)(arow + k0 + half * 8);
        af.p[1] = *(const v8bf*)(arow + k0 + half * 8 + 16);
        #pragma unroll
        for (int j = 0; j < NT; ++j) {
            // B-fragment 32x16, same K order as A: elems 0..7 = K k0+half*8+0..7,
            // elems 8..15 = K k0+16+half*8+0..7 of column l
            union { v16bf v; v8bf p[2]; } bf;
            bf.p[0] = *(const v8bf*)(brow[j] + k0 + half * 8);
            bf.p[1] = *(const v8bf*)(brow[j] + k0 + half * 8 + 16);
            acc[j] = __builtin_amdgcn_wmma_f32_16x16x32_bf16(
                         false, af.v, false, bf.v, (short)0, acc[j], false, false);
        }
    }

    // C/D layout: VGPR r -> row m0 + half*8 + r, col n0 + j*16 + l
    #pragma unroll
    for (int j = 0; j < NT; ++j) {
        float* drow = D + (size_t)(m0 + half * 8) * HC + n0 + j * 16 + l;
        #pragma unroll
        for (int r = 0; r < 8; ++r) drow[(size_t)r * HC] = acc[j][r];
    }
}

// ---------- global mean pool (sums + counts via atomics) ----------
__global__ void k_pool(const float* __restrict__ x, const int* __restrict__ batch,
                       float* __restrict__ pool, float* __restrict__ cnt) {
    long idx = (long)blockIdx.x * blockDim.x + threadIdx.x;
    if (idx >= (long)Nn * HC) return;
    int n = (int)(idx / HC), c = (int)(idx % HC);
    int g = batch[n];
    atomicAdd(&pool[(size_t)g * HC + c], x[idx]);
    if (c == 0) atomicAdd(&cnt[g], 1.0f);
}

// ---------- final: sigmoid( (pool[g]/cnt[g]) . w_lin ) ----------
__global__ void k_final(const float* __restrict__ pool, const float* __restrict__ cnt,
                        const float* __restrict__ wlin, float* __restrict__ out) {
    __shared__ float sh[128];
    int g = blockIdx.x;
    float s = 0.f;
    for (int c = threadIdx.x; c < HC; c += 128)
        s += pool[(size_t)g * HC + c] * wlin[c];
    sh[threadIdx.x] = s;
    __syncthreads();
    for (int st = 64; st > 0; st >>= 1) {
        if (threadIdx.x < st) sh[threadIdx.x] += sh[threadIdx.x + st];
        __syncthreads();
    }
    if (threadIdx.x == 0) {
        float cv = cnt[g]; if (cv < 1.0f) cv = 1.0f;
        float z = sh[0] / cv;
        out[g] = 1.0f / (1.0f + __expf(-z));
    }
}

// =====================================================================
extern "C" void kernel_launch(void* const* d_in, const int* in_sizes, int n_in,
                              void* d_out, int out_size, void* d_ws, size_t ws_size,
                              hipStream_t stream) {
    const float* x        = (const float*)d_in[0];
    const float* ea       = (const float*)d_in[1];
    const int*   eidx     = (const int*)  d_in[2];
    const int*   batch    = (const int*)  d_in[3];
    const float* w1       = (const float*)d_in[4];
    const float* att_src1 = (const float*)d_in[5];
    const float* att_dst1 = (const float*)d_in[6];
    const float* we1      = (const float*)d_in[7];
    const float* atte1    = (const float*)d_in[8];
    const float* b1       = (const float*)d_in[9];
    const float* w2       = (const float*)d_in[10];
    const float* att_src2 = (const float*)d_in[11];
    const float* att_dst2 = (const float*)d_in[12];
    const float* we2      = (const float*)d_in[13];
    const float* atte2    = (const float*)d_in[14];
    const float* b2       = (const float*)d_in[15];
    const float* wlin     = (const float*)d_in[16];
    float* out = (float*)d_out;

    const int* srcI = eidx;        // edge_index[0]
    const int* dstI = eidx + Ee;   // edge_index[1]

    // ---- workspace layout ----
    char* ws = (char*)d_ws;
    size_t off = 0;
    auto alloc = [&](size_t bytes) { size_t r = off; off = (off + bytes + 255) & ~(size_t)255; return r; };
    float*          hbuf  = (float*)         (ws + alloc((size_t)Nn * HC * 4));
    float*          obuf  = (float*)         (ws + alloc((size_t)Nn * HC * 4));
    float*          ealph = (float*)         (ws + alloc((size_t)EP * Hh * 4));
    float*          as_   = (float*)         (ws + alloc((size_t)Nn * Hh * 4));
    float*          ad_   = (float*)         (ws + alloc((size_t)Nn * Hh * 4));
    unsigned*       nmax  = (unsigned*)      (ws + alloc((size_t)Nn * Hh * 4));
    float*          nden  = (float*)         (ws + alloc((size_t)Nn * Hh * 4));
    unsigned short* xbf   = (unsigned short*)(ws + alloc((size_t)Nn * HC * 2));
    unsigned short* wbf   = (unsigned short*)(ws + alloc((size_t)HC * HC * 2));
    float*          pool  = (float*)         (ws + alloc((size_t)Gg * HC * 4));
    float*          cnt   = (float*)         (ws + alloc((size_t)Gg * 4));
    float*          easum = (float*)         (ws + alloc(4));
    float*          coef  = (float*)         (ws + alloc(2 * Hh * 4));
    (void)ws_size; (void)n_in; (void)in_sizes; (void)out_size;

    auto nb = [](long n, int b) { return (unsigned)((n + b - 1) / b); };
    const unsigned NEG_INF_ENC = 0x007FFFFFu;   // enc_f(-inf)

    // ---- preamble: edge-attr mean & per-head edge coefficients ----
    hipMemsetAsync(easum, 0, 4, stream);
    k_mean_sum<<<256, 256, 0, stream>>>(ea, easum);
    k_coef<<<1, 32, 0, stream>>>(we1, atte1, we2, atte2, coef);

    // ================= conv1 =================
    k_proj1<<<nb((long)Nn * HC, 256), 256, 0, stream>>>(x, w1, hbuf);
    k_attn_scores<<<nb((long)Nn * Hh, 256), 256, 0, stream>>>(hbuf, att_src1, att_dst1, as_, ad_);
    k_fill_u32<<<nb((long)Nn * Hh, 256), 256, 0, stream>>>(nmax, NEG_INF_ENC, Nn * Hh);
    hipMemsetAsync(nden, 0, (size_t)Nn * Hh * 4, stream);
    k_edge_alpha<<<nb((long)EP * Hh, 256), 256, 0, stream>>>(srcI, dstI, ea, easum, coef,
                                                             as_, ad_, ealph, nmax);
    k_edge_exp<<<nb((long)EP * Hh, 256), 256, 0, stream>>>(srcI, dstI, ealph, nmax, nden);
    hipMemsetAsync(obuf, 0, (size_t)Nn * HC * 4, stream);
    k_edge_scatter<<<nb((long)EP * Hh * (Cc / 4), 256), 256, 0, stream>>>(srcI, dstI, ealph, nden,
                                                                          hbuf, obuf);
    k_bias_elu<<<nb((long)Nn * HC, 256), 256, 0, stream>>>(obuf, b1);   // obuf = x2

    // ================= conv2 =================
    // bf16 copies for the WMMA GEMM h2 = x2 @ W2^T
    k_f2bf<<<nb((long)Nn * HC, 256), 256, 0, stream>>>(obuf, xbf, (long)Nn * HC);
    k_f2bf<<<nb((long)HC * HC, 256), 256, 0, stream>>>(w2, wbf, (long)HC * HC);
    {
        dim3 grid(Nn / 16, HC / (16 * NT));   // 1250 x 6 strips, one wave each
        k_wmma_gemm<<<grid, 32, 0, stream>>>(xbf, wbf, hbuf);   // hbuf = h2 (f32)
    }
    k_attn_scores<<<nb((long)Nn * Hh, 256), 256, 0, stream>>>(hbuf, att_src2, att_dst2, as_, ad_);
    k_fill_u32<<<nb((long)Nn * Hh, 256), 256, 0, stream>>>(nmax, NEG_INF_ENC, Nn * Hh);
    hipMemsetAsync(nden, 0, (size_t)Nn * Hh * 4, stream);
    k_edge_alpha<<<nb((long)EP * Hh, 256), 256, 0, stream>>>(srcI, dstI, ea, easum, coef + Hh,
                                                             as_, ad_, ealph, nmax);
    k_edge_exp<<<nb((long)EP * Hh, 256), 256, 0, stream>>>(srcI, dstI, ealph, nmax, nden);
    hipMemsetAsync(obuf, 0, (size_t)Nn * HC * 4, stream);   // x2 no longer needed
    k_edge_scatter<<<nb((long)EP * Hh * (Cc / 4), 256), 256, 0, stream>>>(srcI, dstI, ealph, nden,
                                                                          hbuf, obuf);
    k_bias_elu<<<nb((long)Nn * HC, 256), 256, 0, stream>>>(obuf, b2);   // obuf = x4

    // ================= pool + linear + sigmoid =================
    hipMemsetAsync(pool, 0, (size_t)Gg * HC * 4, stream);
    hipMemsetAsync(cnt, 0, (size_t)Gg * 4, stream);
    k_pool<<<nb((long)Nn * HC, 256), 256, 0, stream>>>(obuf, batch, pool, cnt);
    k_final<<<Gg, 128, 0, stream>>>(pool, cnt, wlin, out);
}
